// ProtoNet_45887430591179
// MI455X (gfx1250) — matrix operand order repaired and checked by reading, not compiled, not run on hardware
//
#include <hip/hip_runtime.h>
#include <hip/hip_bf16.h>

typedef __attribute__((ext_vector_type(16))) _Float16 v16h;
typedef __attribute__((ext_vector_type(8)))  _Float16 v8h;
typedef __attribute__((ext_vector_type(4)))  _Float16 v4h;
typedef __attribute__((ext_vector_type(8)))  float    v8f;
typedef __attribute__((ext_vector_type(4)))  float    v4f;

#define P      196
#define PPAD   208           // query rows padded: 13 * 16
#define SPAD   224           // support cols padded per class: 14 * 16 (7 pairs)
#define D      384
#define NC     5
#define MQ     512
#define NTILES 13            // M tiles
#define NPAIRS 7             // N tile pairs per class (14 tiles, last mostly masked)
#define KSTEPS 12            // 384 / 32
#define LDST   (NC * SPAD)   // 1120 columns in transposed support
#define BROW   36            // LDS pair-panel row stride in halfs (32 + 4 pad)
#define NEGINF (-3.0e38f)

// ---------------------------------------------------------------------------
// Phase 1a: L2-normalize query rows -> f16, pad P 196 -> 208.
// ---------------------------------------------------------------------------
__global__ __launch_bounds__(128) void nrm_query(const float* __restrict__ q,
                                                 _Float16* __restrict__ qh) {
  const int wid  = (int)((blockIdx.x * blockDim.x + threadIdx.x) >> 5);
  const int lane = threadIdx.x & 31;
  if (wid >= MQ * PPAD) return;
  const int m = wid / PPAD, p = wid % PPAD;
  _Float16* dst = qh + (size_t)wid * D;
  if (p >= P) {
    v4h z = {};
    for (int j = 0; j < 3; j++) *(v4h*)(dst + 4 * lane + 128 * j) = z;
    return;
  }
  const float* src = q + ((size_t)m * P + p) * D;
  v4f x[3];
  float ss = 0.f;
  for (int j = 0; j < 3; j++) {
    x[j] = *(const v4f*)(src + 4 * lane + 128 * j);
    for (int e = 0; e < 4; e++) ss += x[j][e] * x[j][e];
  }
  for (int off = 16; off > 0; off >>= 1) ss += __shfl_xor(ss, off, 32);
  const float inv = 1.0f / fmaxf(sqrtf(ss), 1e-12f);
  for (int j = 0; j < 3; j++) {
    v4h h;
    for (int e = 0; e < 4; e++) h[e] = (_Float16)(x[j][e] * inv);
    *(v4h*)(dst + 4 * lane + 128 * j) = h;
  }
}

// ---------------------------------------------------------------------------
// Phase 1b: L2-normalize support rows, store TRANSPOSED with 224-col classes:
//   St[k][c*SPAD + p] = normalize(support[c][p])[k]   (f16, pads zeroed)
// ---------------------------------------------------------------------------
__global__ __launch_bounds__(128) void nrm_support(const float* __restrict__ s,
                                                   _Float16* __restrict__ st) {
  const int wid  = (int)((blockIdx.x * blockDim.x + threadIdx.x) >> 5);
  const int lane = threadIdx.x & 31;
  if (wid >= NC * SPAD) return;
  const int c = wid / SPAD, p = wid % SPAD;
  const int col = c * SPAD + p;
  if (p >= P) {
    for (int j = 0; j < 3; j++)
      for (int e = 0; e < 4; e++)
        st[(size_t)(4 * lane + 128 * j + e) * LDST + col] = (_Float16)0.f;
    return;
  }
  const float* src = s + ((size_t)c * P + p) * D;
  v4f x[3];
  float ss = 0.f;
  for (int j = 0; j < 3; j++) {
    x[j] = *(const v4f*)(src + 4 * lane + 128 * j);
    for (int e = 0; e < 4; e++) ss += x[j][e] * x[j][e];
  }
  for (int off = 16; off > 0; off >>= 1) ss += __shfl_xor(ss, off, 32);
  const float inv = 1.0f / fmaxf(sqrtf(ss), 1e-12f);
  for (int j = 0; j < 3; j++)
    for (int e = 0; e < 4; e++)
      st[(size_t)(4 * lane + 128 * j + e) * LDST + col] =
          (_Float16)(x[j][e] * inv);
}

// ---------------------------------------------------------------------------
// Phase 2: one workgroup (8 wave32) per image m.
//   Per class, a 24KB pair-panel (2 N-tiles x 384 K) of St is staged in LDS
//   (double-buffered; copy's global loads issued before the WMMA block, LDS
//   stores after, one barrier per pair). WMMA B fragments come from LDS via
//   conflict-free 16-bit ds loads; the K-loop is software-pipelined with
//   sched_barrier fences so WMMAs wait on only the previous step's (already-complete) loads,
//   not on the just-issued ones.
// ---------------------------------------------------------------------------
__global__ __launch_bounds__(256) void protonet_wmma(
    const _Float16* __restrict__ qh, const _Float16* __restrict__ st,
    const float* __restrict__ scale_cls, const float* __restrict__ bias,
    float* __restrict__ out) {
  __shared__ _Float16 bpan[2][D * BROW];     // 2 x 13824 halfs = 55296 B
  __shared__ float rowmax_lds[NC][PPAD];     // 4160 B

  const int m    = blockIdx.x;
  const int tid  = threadIdx.x;
  const int wave = tid >> 5;
  const int lane = tid & 31;
  const int lh   = lane & 15;   // lane within 16-lane half
  const int hg   = lane >> 4;   // half-group 0/1
  const _Float16* qm = qh + (size_t)m * PPAD * D;

  // copy-lane mapping: chunk = tid + 256*it -> (row, 16B quarter)
  const int crow0 = tid >> 2;        // rows 0..63 at it=0, +64 per it
  const int cq    = (tid & 3) * 8;   // half offset of the 16B quarter

  for (int round = 0; round < 2; round++) {
    const int tile    = wave + 8 * round;
    const bool active = (tile < NTILES);

    // ---- A fragments for this 16-row strip (ISA 16-bit A layout) ----
    v16h A[KSTEPS];
    if (active) {
      const _Float16* qrow = qm + (size_t)(tile * 16 + lh) * D + hg * 8;
#pragma unroll
      for (int k = 0; k < KSTEPS; k++) {
        v8h lo = *(const v8h*)(qrow + k * 32);
        v8h hi = *(const v8h*)(qrow + k * 32 + 16);
        v16h a;
#pragma unroll
        for (int e = 0; e < 8; e++) { a[e] = lo[e]; a[e + 8] = hi[e]; }
        A[k] = a;
      }
    }

    for (int c = 0; c < NC; c++) {
      v8f rmax;
#pragma unroll
      for (int e = 0; e < 8; e++) rmax[e] = NEGINF;

      // ---- prologue: stage pair 0 into buffer 0 ----
      {
        const _Float16* src = st + (size_t)c * SPAD + cq;
#pragma unroll
        for (int it = 0; it < 6; it++) {
          const int row = crow0 + 64 * it;
          v8h g = *(const v8h*)(src + (size_t)row * LDST);
          *(v8h*)(&bpan[0][row * BROW + cq]) = g;
        }
      }
      __syncthreads();

      for (int np = 0; np < NPAIRS; np++) {
        // ---- issue global loads for next pair (overlap with WMMA) ----
        v8h G[6];
        if (np + 1 < NPAIRS) {
          const _Float16* src =
              st + (size_t)c * SPAD + (np + 1) * 32 + cq;
#pragma unroll
          for (int it = 0; it < 6; it++)
            G[it] = *(const v8h*)(src + (size_t)(crow0 + 64 * it) * LDST);
        }

        // ---- compute pair np from LDS (pipelined K loop) ----
        if (active) {
          const _Float16* bp = &bpan[np & 1][0];
          v16h Bc0, Bc1, Bn0, Bn1;
          {  // k = 0 fragments
            const _Float16* p0 = bp + (hg * 8) * BROW + lh;
#pragma unroll
            for (int e = 0; e < 8; e++) {
              Bc0[e] = p0[e * BROW]; Bc0[e + 8] = p0[(16 + e) * BROW];
              Bc1[e] = p0[e * BROW + 16]; Bc1[e + 8] = p0[(16 + e) * BROW + 16];
            }
          }
          v8f acc0 = {0.f, 0.f, 0.f, 0.f, 0.f, 0.f, 0.f, 0.f};
          v8f acc1 = {0.f, 0.f, 0.f, 0.f, 0.f, 0.f, 0.f, 0.f};
#pragma unroll
          for (int k = 0; k < KSTEPS; k++) {
            if (k + 1 < KSTEPS) {         // prefetch next K-step from LDS
              const _Float16* p0 = bp + (32 * (k + 1) + hg * 8) * BROW + lh;
#pragma unroll
              for (int e = 0; e < 8; e++) {
                Bn0[e] = p0[e * BROW]; Bn0[e + 8] = p0[(16 + e) * BROW];
                Bn1[e] = p0[e * BROW + 16]; Bn1[e + 8] = p0[(16 + e) * BROW + 16];
              }
            }
            // fence: loads for k+1 must stay above the k WMMAs
            __builtin_amdgcn_sched_barrier(0);
            acc0 = __builtin_amdgcn_wmma_f32_16x16x32_f16(
                false, A[k], false, Bc0, (short)0, acc0, false, false);
            acc1 = __builtin_amdgcn_wmma_f32_16x16x32_f16(
                false, A[k], false, Bc1, (short)0, acc1, false, false);
            __builtin_amdgcn_sched_barrier(0);
            Bc0 = Bn0;                    // rotate double buffers
            Bc1 = Bn1;
          }
          const bool v0 = (np * 32 + lh) < P;        // sub-tile 2*np
          const bool v1 = (np * 32 + 16 + lh) < P;   // sub-tile 2*np+1
#pragma unroll
          for (int e = 0; e < 8; e++) {
            rmax[e] = fmaxf(rmax[e], v0 ? acc0[e] : NEGINF);
            rmax[e] = fmaxf(rmax[e], v1 ? acc1[e] : NEGINF);
          }
        }

        // ---- store next pair into the other LDS buffer ----
        if (np + 1 < NPAIRS) {
          _Float16* dst = &bpan[(np + 1) & 1][0];
#pragma unroll
          for (int it = 0; it < 6; it++)
            *(v8h*)(dst + (crow0 + 64 * it) * BROW + cq) = G[it];
        }
        __syncthreads();
      }

      if (active) {
        // reduce max across the 16 lanes of each half (C layout: lane = N)
#pragma unroll
        for (int e = 0; e < 8; e++) {
          float v = rmax[e];
          v = fmaxf(v, __shfl_xor(v, 1, 32));
          v = fmaxf(v, __shfl_xor(v, 2, 32));
          v = fmaxf(v, __shfl_xor(v, 4, 32));
          v = fmaxf(v, __shfl_xor(v, 8, 32));
          rmax[e] = v;
        }
        if (lh == 0) {                       // lanes 0 and 16 hold results
          const int rbase = tile * 16 + hg * 8;
#pragma unroll
          for (int e = 0; e < 8; e++) rowmax_lds[c][rbase + e] = rmax[e];
        }
      }
    }
  }
  __syncthreads();

  // ---- top-6 sum over the 196 query-patch maxima, one wave per class ----
  const float sc = scale_cls[0];
  const float bs = bias[0];
  for (int c = wave; c < NC; c += 8) {
    float vals[7];
#pragma unroll
    for (int j = 0; j < 7; j++) {
      const int idx = lane + 32 * j;
      vals[j] = (idx < P) ? rowmax_lds[c][idx] : NEGINF;
    }
    float sum = 0.f;
    for (int it = 0; it < 6; it++) {
      float bm = vals[0];
      int bj = 0;
#pragma unroll
      for (int j = 1; j < 7; j++)
        if (vals[j] > bm) { bm = vals[j]; bj = j; }
      int bi = lane + 32 * bj;
      for (int off = 1; off < 32; off <<= 1) {   // butterfly argmax
        const float ov = __shfl_xor(bm, off, 32);
        const int   oi = __shfl_xor(bi, off, 32);
        if (ov > bm || (ov == bm && oi < bi)) { bm = ov; bi = oi; }
      }
      sum += bm;
      if ((bi & 31) == lane) vals[bi >> 5] = NEGINF;
    }
    if (lane == 0) out[m * NC + c] = sc * (sum + bs);
  }
}

// ---------------------------------------------------------------------------
extern "C" void kernel_launch(void* const* d_in, const int* in_sizes, int n_in,
                              void* d_out, int out_size, void* d_ws,
                              size_t ws_size, hipStream_t stream) {
  const float* support   = (const float*)d_in[0];  // [5, 196, 384]
  const float* query     = (const float*)d_in[1];  // [512, 196, 384]
  const float* scale_cls = (const float*)d_in[2];  // [1]
  const float* bias      = (const float*)d_in[3];  // [1]
  float* out = (float*)d_out;                      // [1, 512, 5] flat

  char* ws = (char*)d_ws;
  _Float16* st = (_Float16*)ws;                    // 384 * 1120 * 2 = 860160 B
  _Float16* qb = (_Float16*)(ws + (size_t)D * LDST * sizeof(_Float16));
                                                   // 512 * 208 * 384 * 2 B

  {
    const int waves = NC * SPAD;                   // 1120
    nrm_support<<<(waves * 32 + 127) / 128, 128, 0, stream>>>(support, st);
  }
  {
    const int waves = MQ * PPAD;                   // 106496
    nrm_query<<<(waves * 32 + 127) / 128, 128, 0, stream>>>(query, qb);
  }
  protonet_wmma<<<MQ, 256, 0, stream>>>(qb, st, scale_cls, bias, out);
}
